// MSK_4544075399268
// MI455X (gfx1250) — compile-verified, matrix operand order checked by reading, not run on hardware
//
#include <hip/hip_runtime.h>

typedef __attribute__((ext_vector_type(16))) _Float16 v16h;
typedef __attribute__((ext_vector_type(8)))  float    v8f;
typedef __attribute__((ext_vector_type(4)))  int      v4i;

#define LAYERS 6
#define BATCH  8
#define SEQ    1024
#define DMODEL 768
#define NHEAD  12
#define DHEAD  64
#define FFDIM  3072
#define MROWS  (BATCH*SEQ)

#if defined(__has_builtin)
#if __has_builtin(__builtin_amdgcn_global_load_async_to_lds_b128)
#define HAVE_ASYNC_LDS 1
#endif
#endif
#ifndef HAVE_ASYNC_LDS
#define HAVE_ASYNC_LDS 0
#endif

#if HAVE_ASYNC_LDS
typedef __attribute__((address_space(1))) v4i gv4i;   // global int4 vector
typedef __attribute__((address_space(3))) v4i lv4i;   // LDS int4 vector
#endif

union F16Frag { v16h v; uint4 q[2]; };

__device__ __forceinline__ v16h ld_frag(const _Float16* p0, const _Float16* p1) {
    F16Frag f;
    f.q[0] = *(const uint4*)p0;
    f.q[1] = *(const uint4*)p1;
    return f.v;
}

// 16-byte global -> LDS copy; async (ASYNCcnt) path on CDNA5 toolchains.
__device__ __forceinline__ void cp16_g2l(const _Float16* g, _Float16* l) {
#if HAVE_ASYNC_LDS
    __builtin_amdgcn_global_load_async_to_lds_b128((gv4i*)g, (lv4i*)l, 0, 0);
#else
    *(uint4*)l = *(const uint4*)g;
#endif
}

__device__ __forceinline__ void wait_async0() {
#if HAVE_ASYNC_LDS
#if __has_builtin(__builtin_amdgcn_s_wait_asynccnt)
    __builtin_amdgcn_s_wait_asynccnt(0);
#else
    asm volatile("s_wait_asynccnt 0" ::: "memory");
#endif
#endif
}

// ---------------------------------------------------------------------------
// Embedding: h[b,s,:] = wte[id] + wpe[s]
// ---------------------------------------------------------------------------
__global__ void embed_k(const int* __restrict__ ids, const float* __restrict__ wte,
                        const float* __restrict__ wpe, float* __restrict__ h) {
    int row = blockIdx.x;
    int s   = row & (SEQ - 1);
    int id  = ids[row];
    for (int d = threadIdx.x; d < DMODEL; d += blockDim.x)
        h[(size_t)row * DMODEL + d] = wte[(size_t)id * DMODEL + d] + wpe[(size_t)s * DMODEL + d];
}

// ---------------------------------------------------------------------------
// LayerNorm over D=768, one block per row
// ---------------------------------------------------------------------------
template<bool O16, bool O32>
__global__ __launch_bounds__(256) void ln_fwd(const float* __restrict__ x,
                                              const float* __restrict__ g,
                                              const float* __restrict__ b,
                                              _Float16* o16, float* o32) {
    __shared__ float red[256];
    const int row = blockIdx.x, tid = threadIdx.x;
    const float* xr = x + (size_t)row * DMODEL;
    float l[3], s = 0.f;
    #pragma unroll
    for (int j = 0; j < 3; ++j) { l[j] = xr[tid + j * 256]; s += l[j]; }
    red[tid] = s; __syncthreads();
    for (int off = 128; off > 0; off >>= 1) {
        if (tid < off) red[tid] += red[tid + off];
        __syncthreads();
    }
    const float mean = red[0] * (1.0f / DMODEL);
    __syncthreads();
    float s2 = 0.f;
    #pragma unroll
    for (int j = 0; j < 3; ++j) { float d = l[j] - mean; s2 += d * d; }
    red[tid] = s2; __syncthreads();
    for (int off = 128; off > 0; off >>= 1) {
        if (tid < off) red[tid] += red[tid + off];
        __syncthreads();
    }
    const float rstd = rsqrtf(red[0] * (1.0f / DMODEL) + 1e-5f);
    #pragma unroll
    for (int j = 0; j < 3; ++j) {
        int c = tid + j * 256;
        float v = (l[j] - mean) * rstd * g[c] + b[c];
        if (O16) o16[(size_t)row * DMODEL + c] = (_Float16)v;
        if (O32) o32[(size_t)row * DMODEL + c] = v;
    }
}

// ---------------------------------------------------------------------------
// f32 [K,N] -> f16 transposed [N,K], 32x32 LDS tiles (both sides coalesced)
// ---------------------------------------------------------------------------
__global__ __launch_bounds__(256) void cvt_t_f16(const float* __restrict__ in,
                                                 _Float16* __restrict__ out,
                                                 int K, int N) {
    __shared__ float t[32][33];
    const int tx = threadIdx.x & 31, ty = threadIdx.x >> 5;   // 32 x 8
    const int k0 = blockIdx.y * 32, n0 = blockIdx.x * 32;
    #pragma unroll
    for (int j = 0; j < 4; ++j)
        t[ty + j * 8][tx] = in[(size_t)(k0 + ty + j * 8) * N + n0 + tx];
    __syncthreads();
    #pragma unroll
    for (int j = 0; j < 4; ++j)
        out[(size_t)(n0 + ty + j * 8) * K + k0 + tx] = (_Float16)t[tx][ty + j * 8];
}

// ---------------------------------------------------------------------------
// GEMM (TN): C[M,N] = A[M,K](f16,rm) @ Bt[N,K](f16,rm)^T + bias (+resid)(+gelu)
// 128x128 block tile, 8 waves (4x2), K-stage 64, double-buffered LDS with
// async global->LDS staging; v_wmma_f32_16x16x32_f16 main loop.
// ---------------------------------------------------------------------------
template<bool RESID, bool GELU, bool OUTF, bool OUTH>
__global__ __launch_bounds__(256) void gemm_tn(const _Float16* __restrict__ A,
                                               const _Float16* __restrict__ Bt,
                                               const float* __restrict__ bias,
                                               const float* resid, float* Cf,
                                               _Float16* Ch, int M, int N, int K) {
    __shared__ _Float16 As[2][128 * 72];
    __shared__ _Float16 Bs[2][128 * 72];
    const int tid = threadIdx.x;
    const int wid = tid >> 5, lane = tid & 31;
    const int lm = lane & 15, lh = lane >> 4;
    const int wm = wid & 3, wn = wid >> 2;
    const int m0 = blockIdx.y * 128, n0 = blockIdx.x * 128;

    v8f acc[2][4];
    #pragma unroll
    for (int i = 0; i < 2; ++i)
        #pragma unroll
        for (int j = 0; j < 4; ++j)
            #pragma unroll
            for (int e = 0; e < 8; ++e) acc[i][j][e] = 0.f;

    auto stage = [&](int buf, int k0) {
        // 128 rows x 64 halves each for A and Bt: 1024 16B chunks apiece
        #pragma unroll
        for (int p = 0; p < 4; ++p) {
            int cc = tid + p * 256;
            int r = cc >> 3, kc = (cc & 7) * 8;
            cp16_g2l(A  + (size_t)(m0 + r) * K + k0 + kc, &As[buf][r * 72 + kc]);
            cp16_g2l(Bt + (size_t)(n0 + r) * K + k0 + kc, &Bs[buf][r * 72 + kc]);
        }
    };

    stage(0, 0);
    wait_async0();
    __syncthreads();

    int buf = 0;
    for (int k0 = 0; k0 < K; k0 += 64) {
        if (k0 + 64 < K) stage(buf ^ 1, k0 + 64);   // prefetch next stage
        #pragma unroll
        for (int ks = 0; ks < 2; ++ks) {
            v16h afrag[2];
            #pragma unroll
            for (int mt = 0; mt < 2; ++mt) {
                int r = wm * 32 + mt * 16 + lm;
                afrag[mt] = ld_frag(&As[buf][r * 72 + ks * 32 + lh * 8],
                                    &As[buf][r * 72 + ks * 32 + 16 + lh * 8]);
            }
            #pragma unroll
            for (int nt = 0; nt < 4; ++nt) {
                int c = wn * 64 + nt * 16 + lm;
                v16h bfrag = ld_frag(&Bs[buf][c * 72 + ks * 32 + lh * 8],
                                     &Bs[buf][c * 72 + ks * 32 + 16 + lh * 8]);
                #pragma unroll
                for (int mt = 0; mt < 2; ++mt)
                    acc[mt][nt] = __builtin_amdgcn_wmma_f32_16x16x32_f16(
                        false, afrag[mt], false, bfrag, (short)0, acc[mt][nt],
                        false, false);
            }
        }
        wait_async0();
        __syncthreads();
        buf ^= 1;
    }

    const float GC = 0.7978845608028654f;
    #pragma unroll
    for (int mt = 0; mt < 2; ++mt)
        #pragma unroll
        for (int nt = 0; nt < 4; ++nt)
            #pragma unroll
            for (int i = 0; i < 8; ++i) {
                int r = m0 + wm * 32 + mt * 16 + i + lh * 8;
                int c = n0 + wn * 64 + nt * 16 + lm;
                size_t idx = (size_t)r * N + c;
                float v = acc[mt][nt][i] + bias[c];
                if (RESID) v += resid[idx];
                if (GELU) {
                    float u = GC * (v + 0.044715f * v * v * v);
                    v = 0.5f * v * (1.0f + tanhf(u));
                }
                if (OUTF) Cf[idx] = v;
                if (OUTH) Ch[idx] = (_Float16)v;
            }
}

// ---------------------------------------------------------------------------
// Flash attention: one block per (b, h, 128-row q tile); online softmax;
// QK^T and PV via WMMA; causal + calibrated additive mask; out f16 [B*S, D]
// ---------------------------------------------------------------------------
__global__ __launch_bounds__(256) void attn_k(const float* __restrict__ qkv,
                                              const float* __restrict__ mask,
                                              _Float16* __restrict__ out16) {
    __shared__ _Float16 Qs[128 * 72];
    __shared__ _Float16 Ks[32 * 72];
    __shared__ _Float16 Vt[64 * 40];
    __shared__ _Float16 Ps[128 * 40];
    const int tid = threadIdx.x;
    const int wid = tid >> 5, lane = tid & 31;
    const int lm = lane & 15, lh = lane >> 4;
    const int q0 = blockIdx.x * 128, hh = blockIdx.y, bb = blockIdx.z;
    const size_t rowbase = (size_t)bb * SEQ;
    const int W3 = 3 * DMODEL;
    const int qoff = hh * DHEAD, koff = DMODEL + hh * DHEAD, voff = 2 * DMODEL + hh * DHEAD;

    #pragma unroll
    for (int p = 0; p < 8; ++p) {
        int cc = tid + p * 256;
        int r = cc >> 4, d0 = (cc & 15) * 4;
        float4 f = *(const float4*)(qkv + (rowbase + q0 + r) * W3 + qoff + d0);
        Qs[r * 72 + d0 + 0] = (_Float16)f.x;
        Qs[r * 72 + d0 + 1] = (_Float16)f.y;
        Qs[r * 72 + d0 + 2] = (_Float16)f.z;
        Qs[r * 72 + d0 + 3] = (_Float16)f.w;
    }

    float mprev[8], lsum[8];
    v8f accO[4];
    #pragma unroll
    for (int i = 0; i < 8; ++i) { mprev[i] = -3.0e38f; lsum[i] = 0.f; }
    #pragma unroll
    for (int j = 0; j < 4; ++j)
        #pragma unroll
        for (int e = 0; e < 8; ++e) accO[j][e] = 0.f;

    const int nkt = q0 / 32 + 4;
    for (int kt = 0; kt < nkt; ++kt) {
        __syncthreads();
        #pragma unroll
        for (int p = 0; p < 2; ++p) {
            int cc = tid + p * 256;
            int kr = cc >> 4, d0 = (cc & 15) * 4;
            const size_t grow = (rowbase + kt * 32 + kr) * W3;
            float4 f = *(const float4*)(qkv + grow + koff + d0);
            Ks[kr * 72 + d0 + 0] = (_Float16)f.x;
            Ks[kr * 72 + d0 + 1] = (_Float16)f.y;
            Ks[kr * 72 + d0 + 2] = (_Float16)f.z;
            Ks[kr * 72 + d0 + 3] = (_Float16)f.w;
            float4 g = *(const float4*)(qkv + grow + voff + d0);
            Vt[(d0 + 0) * 40 + kr] = (_Float16)g.x;
            Vt[(d0 + 1) * 40 + kr] = (_Float16)g.y;
            Vt[(d0 + 2) * 40 + kr] = (_Float16)g.z;
            Vt[(d0 + 3) * 40 + kr] = (_Float16)g.w;
        }
        __syncthreads();

        v8f sacc[2];
        #pragma unroll
        for (int nt = 0; nt < 2; ++nt)
            #pragma unroll
            for (int e = 0; e < 8; ++e) sacc[nt][e] = 0.f;
        #pragma unroll
        for (int ks = 0; ks < 2; ++ks) {
            int r = wid * 16 + lm;
            v16h aq = ld_frag(&Qs[r * 72 + ks * 32 + lh * 8],
                              &Qs[r * 72 + ks * 32 + 16 + lh * 8]);
            #pragma unroll
            for (int nt = 0; nt < 2; ++nt) {
                int n = nt * 16 + lm;
                v16h bk = ld_frag(&Ks[n * 72 + ks * 32 + lh * 8],
                                  &Ks[n * 72 + ks * 32 + 16 + lh * 8]);
                sacc[nt] = __builtin_amdgcn_wmma_f32_16x16x32_f16(
                    false, aq, false, bk, (short)0, sacc[nt], false, false);
            }
        }

        #pragma unroll
        for (int i = 0; i < 8; ++i) {
            const int qrow = q0 + wid * 16 + i + lh * 8;
            float s0 = sacc[0][i] * 0.125f;
            {
                int kc = kt * 32 + lm;
                if (kc > qrow) s0 = -1.0e30f;
                else s0 += mask[(rowbase + qrow) * SEQ + kc];
            }
            float s1 = sacc[1][i] * 0.125f;
            {
                int kc = kt * 32 + 16 + lm;
                if (kc > qrow) s1 = -1.0e30f;
                else s1 += mask[(rowbase + qrow) * SEQ + kc];
            }
            float mx = fmaxf(s0, s1);
            mx = fmaxf(mx, __shfl_xor(mx, 1, 32));
            mx = fmaxf(mx, __shfl_xor(mx, 2, 32));
            mx = fmaxf(mx, __shfl_xor(mx, 4, 32));
            mx = fmaxf(mx, __shfl_xor(mx, 8, 32));
            const float mnew = fmaxf(mprev[i], mx);
            const float p0 = __expf(s0 - mnew);
            const float p1 = __expf(s1 - mnew);
            float rs = p0 + p1;
            rs += __shfl_xor(rs, 1, 32);
            rs += __shfl_xor(rs, 2, 32);
            rs += __shfl_xor(rs, 4, 32);
            rs += __shfl_xor(rs, 8, 32);
            const float alpha = __expf(mprev[i] - mnew);
            lsum[i] = lsum[i] * alpha + rs;
            mprev[i] = mnew;
            #pragma unroll
            for (int j = 0; j < 4; ++j) accO[j][i] *= alpha;
            const int pr = wid * 16 + i + lh * 8;
            Ps[pr * 40 + lm]      = (_Float16)p0;
            Ps[pr * 40 + 16 + lm] = (_Float16)p1;
        }

        {
            int r = wid * 16 + lm;
            v16h pa = ld_frag(&Ps[r * 40 + lh * 8], &Ps[r * 40 + 16 + lh * 8]);
            #pragma unroll
            for (int nt = 0; nt < 4; ++nt) {
                int n = nt * 16 + lm;
                v16h vb = ld_frag(&Vt[n * 40 + lh * 8], &Vt[n * 40 + 16 + lh * 8]);
                accO[nt] = __builtin_amdgcn_wmma_f32_16x16x32_f16(
                    false, pa, false, vb, (short)0, accO[nt], false, false);
            }
        }
    }

    #pragma unroll
    for (int nt = 0; nt < 4; ++nt)
        #pragma unroll
        for (int i = 0; i < 8; ++i) {
            const int qrow = q0 + wid * 16 + i + lh * 8;
            const int c = hh * DHEAD + nt * 16 + lm;
            float v = accO[nt][i] / lsum[i];
            out16[(rowbase + qrow) * DMODEL + c] = (_Float16)v;
        }
}

// ---------------------------------------------------------------------------
extern "C" void kernel_launch(void* const* d_in, const int* in_sizes, int n_in,
                              void* d_out, int out_size, void* d_ws, size_t ws_size,
                              hipStream_t stream) {
    const int*   x_ids = (const int*)  d_in[0];
    const float* cmask = (const float*)d_in[1];
    const float* wte   = (const float*)d_in[2];
    const float* wpe   = (const float*)d_in[3];
    const float* ln1_g = (const float*)d_in[4];
    const float* ln1_b = (const float*)d_in[5];
    const float* qkv_w = (const float*)d_in[6];
    const float* qkv_b = (const float*)d_in[7];
    const float* ao_w  = (const float*)d_in[8];
    const float* ao_b  = (const float*)d_in[9];
    const float* ln2_g = (const float*)d_in[10];
    const float* ln2_b = (const float*)d_in[11];
    const float* fc_w  = (const float*)d_in[12];
    const float* fc_b  = (const float*)d_in[13];
    const float* op_w  = (const float*)d_in[14];
    const float* op_b  = (const float*)d_in[15];
    const float* lnf_g = (const float*)d_in[16];
    const float* lnf_b = (const float*)d_in[17];

    char* ws = (char*)d_ws;
    size_t off = 0;
    auto alloc = [&](size_t bytes) -> char* {
        char* p = ws + off;
        off += (bytes + 255) & ~(size_t)255;
        return p;
    };
    float*    h    = (float*)   alloc((size_t)MROWS * DMODEL * 4);
    _Float16* x16  = (_Float16*)alloc((size_t)MROWS * DMODEL * 2);   // LN out / attn out
    _Float16* w16  = (_Float16*)alloc((size_t)DMODEL * FFDIM * 2);   // f16 weight^T staging
    float*    qkv  = (float*)   alloc((size_t)MROWS * 3 * DMODEL * 4);
    _Float16* ff16 = (_Float16*)alloc((size_t)MROWS * FFDIM * 2);

    embed_k<<<MROWS, 256, 0, stream>>>(x_ids, wte, wpe, h);

    for (int l = 0; l < LAYERS; ++l) {
        // LN1 -> f16 activations
        ln_fwd<true, false><<<MROWS, 256, 0, stream>>>(
            h, ln1_g + l * DMODEL, ln1_b + l * DMODEL, x16, nullptr);
        // qkv = x @ qkv_w + qkv_b  (f32 out)
        cvt_t_f16<<<dim3(3 * DMODEL / 32, DMODEL / 32), 256, 0, stream>>>(
            qkv_w + (size_t)l * DMODEL * 3 * DMODEL, w16, DMODEL, 3 * DMODEL);
        gemm_tn<false, false, true, false>
            <<<dim3(3 * DMODEL / 128, MROWS / 128), 256, 0, stream>>>(
            x16, w16, qkv_b + l * 3 * DMODEL, nullptr, qkv, nullptr,
            MROWS, 3 * DMODEL, DMODEL);
        // attention -> f16 [B*S, D]
        attn_k<<<dim3(SEQ / 128, NHEAD, BATCH), 256, 0, stream>>>(qkv, cmask, x16);
        // h = h + attn @ ao_w + ao_b
        cvt_t_f16<<<dim3(DMODEL / 32, DMODEL / 32), 256, 0, stream>>>(
            ao_w + (size_t)l * DMODEL * DMODEL, w16, DMODEL, DMODEL);
        gemm_tn<true, false, true, false>
            <<<dim3(DMODEL / 128, MROWS / 128), 256, 0, stream>>>(
            x16, w16, ao_b + l * DMODEL, h, h, nullptr, MROWS, DMODEL, DMODEL);
        // LN2 -> f16
        ln_fwd<true, false><<<MROWS, 256, 0, stream>>>(
            h, ln2_g + l * DMODEL, ln2_b + l * DMODEL, x16, nullptr);
        // ff = gelu(x @ fc_w + fc_b)  (f16 out)
        cvt_t_f16<<<dim3(FFDIM / 32, DMODEL / 32), 256, 0, stream>>>(
            fc_w + (size_t)l * DMODEL * FFDIM, w16, DMODEL, FFDIM);
        gemm_tn<false, true, false, true>
            <<<dim3(FFDIM / 128, MROWS / 128), 256, 0, stream>>>(
            x16, w16, fc_b + l * FFDIM, nullptr, nullptr, ff16,
            MROWS, FFDIM, DMODEL);
        // h = h + ff @ op_w + op_b
        cvt_t_f16<<<dim3(DMODEL / 32, FFDIM / 32), 256, 0, stream>>>(
            op_w + (size_t)l * FFDIM * DMODEL, w16, FFDIM, DMODEL);
        gemm_tn<true, false, true, false>
            <<<dim3(DMODEL / 128, MROWS / 128), 256, 0, stream>>>(
            ff16, w16, op_b + l * DMODEL, h, h, nullptr, MROWS, DMODEL, FFDIM);
    }

    ln_fwd<false, true><<<MROWS, 256, 0, stream>>>(
        h, lnf_g, lnf_b, nullptr, (float*)d_out);
}
